// Model_43413529428458
// MI455X (gfx1250) — matrix operand order repaired and checked by reading, not compiled, not run on hardware
//
#include <hip/hip_runtime.h>
#include <math.h>
#include <stdint.h>

typedef __attribute__((ext_vector_type(16))) _Float16 v16h;
typedef __attribute__((ext_vector_type(8)))  float    v8f;
typedef __attribute__((ext_vector_type(8)))  _Float16 h8;

namespace {

constexpr int NB  = 256;   // batch == graph nodes
constexpr int NT  = 128;   // timesteps
constexpr int XDM = 128;
constexpr int DGD = 64;
constexpr int DMD = 32;
constexpr int DDM = 224;   // XDM+DGD+DMD
constexpr int HID = 256;
constexpr int C0  = 704;   // 2*DDM + HID
constexpr int GIN = 2816;  // 4*C0

__device__ __forceinline__ float sigf(float x){ return 1.0f/(1.0f+__expf(-x)); }

// A operand: row-major [M,K] f16, ISA 16x32 A striping.
__device__ __forceinline__ v16h load_a16(const _Float16* base, int lda, int lane){
  int m = lane & 15;
  int g = lane >> 4;
  const _Float16* p = base + (long)m*lda + g*8;
  h8 lo = *(const h8*)p;
  h8 hi = *(const h8*)(p + 16);
  v16h r;
#pragma unroll
  for (int i=0;i<8;i++){ r[i]=lo[i]; r[i+8]=hi[i]; }
  return r;
}
// B operand: row-major [K,N] f16; lane%16 = n, elements = k in A's table order.
__device__ __forceinline__ v16h load_b16(const _Float16* base, int ldb, int lane){
  int n = lane & 15;
  int g = lane >> 4;
  v16h r;
#pragma unroll
  for (int i=0;i<8;i++){ r[i]=base[(long)(g*8+i)*ldb+n]; r[i+8]=base[(long)(16+g*8+i)*ldb+n]; }
  return r;
}
__device__ __forceinline__ v8f wmma16(v16h a, v16h b, v8f c){
  return __builtin_amdgcn_wmma_f32_16x16x32_f16(false, a, false, b, (short)0, c, false, false);
}

// ---------------- generic WMMA GEMM, 32x32 macro-tile per wave ----------------
// C[M,N] = A[M,K] @ B[K,N] (+bias, activation). M,N multiples of 32, K of 32.
// biasMode: 0 none, 1 per-row, 2 per-col. act: 0 none, 1 sigmoid, 2 tanh, 3 exp(-relu).
// If Ch != null write f16 else f32. blockIdx.y batches B and C.
__global__ void k_gemm(const _Float16* __restrict__ A, int lda,
                       const _Float16* __restrict__ Bm, int ldb,
                       float* __restrict__ Cf, _Float16* __restrict__ Ch, int ldc,
                       int M, int N, int K,
                       const float* __restrict__ bias, int biasMode, int act,
                       long strideB, long strideC)
{
  int lane = threadIdx.x & 31;
  int wave = threadIdx.x >> 5;
  int tile = blockIdx.x * (blockDim.x >> 5) + wave;
  int tn = N >> 5;
  if (tile >= (M >> 5) * tn) return;
  int m0 = (tile / tn) << 5;
  int n0 = (tile % tn) << 5;
  long bi = blockIdx.y;
  const _Float16* Ap0 = A + (long)m0 * lda;
  const _Float16* Ap1 = Ap0 + (long)16 * lda;
  const _Float16* Bp  = Bm + bi * strideB + n0;
  v8f cc[2][2] = {{{}, {}}, {{}, {}}};
  for (int k0 = 0; k0 < K; k0 += 32){
    v16h a0 = load_a16(Ap0 + k0, lda, lane);
    v16h a1 = load_a16(Ap1 + k0, lda, lane);
    const _Float16* bp = Bp + (long)k0 * ldb;
    v16h b0 = load_b16(bp,      ldb, lane);
    v16h b1 = load_b16(bp + 16, ldb, lane);
    cc[0][0] = wmma16(a0, b0, cc[0][0]);
    cc[0][1] = wmma16(a0, b1, cc[0][1]);
    cc[1][0] = wmma16(a1, b0, cc[1][0]);
    cc[1][1] = wmma16(a1, b1, cc[1][1]);
  }
#pragma unroll
  for (int ti = 0; ti < 2; ti++)
#pragma unroll
    for (int tj = 0; tj < 2; tj++){
      int mrow = m0 + ti*16 + 8*(lane >> 4);
      int n    = n0 + tj*16 + (lane & 15);
#pragma unroll
      for (int r = 0; r < 8; r++){
        int mm = mrow + r;
        float v = cc[ti][tj][r];
        if (biasMode == 1) v += bias[mm];
        else if (biasMode == 2) v += bias[n];
        if (act == 1) v = sigf(v);
        else if (act == 2) v = tanhf(v);
        else if (act == 3) v = __expf(-fmaxf(v, 0.0f));
        long off = bi * strideC + (long)mm * ldc + n;
        if (Ch) Ch[off] = (_Float16)v;
        else    Cf[off] = v;
      }
    }
}

// ------- big per-step GEMM: A[M,704] @ Z0[704,256] -> f16, async B panel in LDS -------
// Block = 8 waves; block tile = 256(M) x 32(N). B panel [704,32] staged via
// GLOBAL_LOAD_ASYNC_TO_LDS_B128 (ASYNCcnt), then 2x2 register-tiled WMMA.
__global__ void k_gemm_big(const _Float16* __restrict__ A,
                           const _Float16* __restrict__ Bz,
                           _Float16* __restrict__ Ch)
{
  __shared__ _Float16 panel[C0][32];   // 45056 B
  int tid  = threadIdx.x;              // 256
  int lane = tid & 31, wave = tid >> 5;
  int n0 = blockIdx.y << 5;
  unsigned ldsBase = (unsigned)(uintptr_t)(&panel[0][0]);
  // stage 704x32 halves = 2816 x 16B transfers; LDS byte off = tau*16 matches [k][32] layout
  for (int tau = tid; tau < C0*4; tau += 256){
    const void* src = (const void*)(Bz + (long)(tau >> 2)*NB + n0 + (tau & 3)*8);
    unsigned dst = ldsBase + (unsigned)tau*16u;
    asm volatile("global_load_async_to_lds_b128 %0, %1, off" :: "v"(dst), "v"(src) : "memory");
  }
  asm volatile("s_wait_asynccnt 0x0" ::: "memory");
  __syncthreads();
  int m0 = (blockIdx.x << 8) + (wave << 5);
  const _Float16* Ap0 = A + (long)m0 * C0;
  const _Float16* Ap1 = Ap0 + (long)16 * C0;
  v8f cc[2][2] = {{{}, {}}, {{}, {}}};
  for (int k0 = 0; k0 < C0; k0 += 32){
    v16h a0 = load_a16(Ap0 + k0, C0, lane);
    v16h a1 = load_a16(Ap1 + k0, C0, lane);
    v16h b0 = load_b16(&panel[k0][0],  32, lane);
    v16h b1 = load_b16(&panel[k0][16], 32, lane);
    cc[0][0] = wmma16(a0, b0, cc[0][0]);
    cc[0][1] = wmma16(a0, b1, cc[0][1]);
    cc[1][0] = wmma16(a1, b0, cc[1][0]);
    cc[1][1] = wmma16(a1, b1, cc[1][1]);
  }
#pragma unroll
  for (int ti = 0; ti < 2; ti++)
#pragma unroll
    for (int tj = 0; tj < 2; tj++){
      int mrow = m0 + ti*16 + 8*(lane >> 4);
      int n    = n0 + tj*16 + (lane & 15);
#pragma unroll
      for (int r = 0; r < 8; r++)
        Ch[(long)(mrow + r)*NB + n] = (_Float16)cc[ti][tj][r];
    }
}

// ---------------- small utility kernels ----------------
__global__ void k_zero(unsigned int* p, long n){
  long i = (long)blockIdx.x * blockDim.x + threadIdx.x;
  if (i < n) p[i] = 0u;
}
__global__ void k_c16(const float* in, _Float16* out, long n){
  long i = (long)blockIdx.x * blockDim.x + threadIdx.x;
  if (i < n) out[i] = (_Float16)in[i];
}
// out f16 [C][R]: out[c*R+r] = in[r*C+c]   (builds W^T for B operands)
__global__ void k_t16(const float* in, _Float16* out, int R, int C){
  long i = (long)blockIdx.x * blockDim.x + threadIdx.x;
  if (i >= (long)R * C) return;
  int r = (int)(i / C), c = (int)(i % C);
  out[(long)c * R + r] = (_Float16)in[i];
}

// ---------------- imputation ----------------
__global__ void k_imp_stats(const float* x, float* t_mean, float* s_acc){
  int b = blockIdx.x, f = threadIdx.x; // 128 threads
  float s = 0.f, c = 0.f;
  for (int t = 0; t < NT; t++){
    float v = x[((long)b * NT + t) * XDM + f];
    if (v != -1.0f){ s += v; c += 1.0f; }
  }
  t_mean[b * XDM + f] = s / (c + 1e-8f);
  atomicAdd(&s_acc[f], s);
  atomicAdd(&s_acc[XDM + f], c);
}
__global__ void k_imp_fill(const float* x, const float* t_mean, const float* s_acc,
                           _Float16* xf16){
  long i = (long)blockIdx.x * blockDim.x + threadIdx.x;
  if (i >= (long)NB * NT * XDM) return;
  int f = (int)(i % XDM);
  int b = (int)(i / ((long)NT * XDM));
  float v = x[i];
  if (v == -1.0f){
    float sm = s_acc[f] / (s_acc[XDM + f] + 1e-8f);
    v = 0.5f * t_mean[b * XDM + f] + 0.5f * sm;
  }
  xf16[i] = (_Float16)v;
}

// ---------------- persistent GRU: 16 batch rows per block, loops all T ----------------
__global__ void k_gru(const _Float16* __restrict__ xf16,
                      const _Float16* __restrict__ WihT, const _Float16* __restrict__ WhhT,
                      const float* __restrict__ bih, const float* __restrict__ bhh,
                      const float* __restrict__ h0, float* __restrict__ gru_out){
  __shared__ float    hS[16][XDM];
  __shared__ _Float16 hH[16][XDM];
  __shared__ _Float16 xS[16][XDM];
  __shared__ float    gS[16][768];     // [gi(384) | gh(384)]
  int tid = threadIdx.x;               // 256 threads = 8 waves
  int lane = tid & 31, wave = tid >> 5;
  int b0 = blockIdx.x * 16;
  {
    int row = tid >> 4, f0 = (tid & 15) * 8;
#pragma unroll
    for (int e = 0; e < 8; e++){
      float v = h0[f0 + e];
      hS[row][f0 + e] = v; hH[row][f0 + e] = (_Float16)v;
    }
  }
  __syncthreads();
  for (int t = 0; t < NT; t++){
    { int row = tid >> 4, f0 = (tid & 15) * 8;
      const _Float16* src = xf16 + ((long)(b0 + row) * NT + t) * XDM + f0;
#pragma unroll
      for (int e = 0; e < 8; e++) xS[row][f0 + e] = src[e];
    }
    __syncthreads();
    for (int j = wave; j < 48; j += 8){            // 24 gi tiles + 24 gh tiles
      bool gi = j < 24;
      int col = gi ? j : (j - 24);
      const _Float16* W = gi ? WihT : WhhT;
      const _Float16* Asrc = gi ? &xS[0][0] : &hH[0][0];
      v8f acc = {};
      for (int k0 = 0; k0 < XDM; k0 += 32){
        v16h a = load_a16(Asrc + k0, XDM, lane);
        v16h b = load_b16(W + (long)k0 * 384 + col * 16, 384, lane);
        acc = wmma16(a, b, acc);
      }
      int mr = 8 * (lane >> 4);
      int cb = (gi ? 0 : 384) + col * 16 + (lane & 15);
#pragma unroll
      for (int r = 0; r < 8; r++) gS[mr + r][cb] = acc[r];
    }
    __syncthreads();
    { int row = tid >> 4, j0 = (tid & 15) * 8;
#pragma unroll
      for (int e = 0; e < 8; e++){
        int j = j0 + e;
        float ir = gS[row][j]         + bih[j];
        float iz = gS[row][128 + j]   + bih[128 + j];
        float in_= gS[row][256 + j]   + bih[256 + j];
        float hr = gS[row][384 + j]       + bhh[j];
        float hz = gS[row][384 + 128 + j] + bhh[128 + j];
        float hn = gS[row][384 + 256 + j] + bhh[256 + j];
        float r  = sigf(ir + hr);
        float z  = sigf(iz + hz);
        float nn = tanhf(in_ + r * hn);
        float h  = (1.0f - z) * nn + z * hS[row][j];
        hS[row][j] = h; hH[row][j] = (_Float16)h;
        gru_out[((long)(b0 + row) * NT + t) * XDM + j] = h;
      }
    }
    __syncthreads();
  }
}

__global__ void k_last(const float* gru_out, const int* lens, _Float16* last16){
  int b = blockIdx.x, f = threadIdx.x; // 128
  int t = lens[b] - 1;
  last16[(long)b * XDM + f] = (_Float16)gru_out[((long)b * NT + t) * XDM + f];
}

// ---------------- attention adjacency (scores + softmax) ----------------
__global__ void k_attn(const float* q, const float* kmat, const float* Wo, const float* bo,
                       int dim, int nh, float* adjf, _Float16* adjall, int colOff){
  __shared__ float qw[128];
  __shared__ float red[256];
  int b = blockIdx.x, tid = threadIdx.x; // 256 threads
  int dk = dim / nh;
  if (tid < dim) qw[tid] = q[(long)b * dim + tid] * Wo[tid / dk] * rsqrtf((float)dk);
  __syncthreads();
  float s = bo[0];
  const float* kr = kmat + (long)tid * dim;
  for (int d = 0; d < dim; d++) s += qw[d] * kr[d];
  red[tid] = s; __syncthreads();
  for (int o = 128; o > 0; o >>= 1){ if (tid < o) red[tid] = fmaxf(red[tid], red[tid + o]); __syncthreads(); }
  float mx = red[0]; __syncthreads();
  float e = __expf(s - mx);
  red[tid] = e; __syncthreads();
  for (int o = 128; o > 0; o >>= 1){ if (tid < o) red[tid] += red[tid + o]; __syncthreads(); }
  float a = e / red[0];
  adjf[(long)b * NB + tid] = a;
  adjall[(long)b * 768 + colOff + tid] = (_Float16)a;
}

__global__ void k_fuse(const float* q_all, const float* q_ts, const float* q_di, const float* q_de,
                       const float* a_ts, const float* a_di, const float* a_de,
                       float* adjF, _Float16* adjF16){
  __shared__ float qa[128];
  int b = blockIdx.x, c = threadIdx.x;
  if (c < 128) qa[c] = q_all[(long)b * 128 + c];
  __syncthreads();
  float s0 = 0, s1 = 0, s2 = 0;
  const float* p0 = q_ts + (long)c * 128;
  const float* p1 = q_di + (long)c * 128;
  const float* p2 = q_de + (long)c * 128;
  for (int d = 0; d < 128; d++){
    float qv = qa[d];
    s0 += qv * p0[d]; s1 += qv * p1[d]; s2 += qv * p2[d];
  }
  const float inv = 0.08838834764831845f; // 1/sqrt(128)
  s0 *= inv; s1 *= inv; s2 *= inv;
  float m = fmaxf(s0, fmaxf(s1, s2));
  float e0 = __expf(s0 - m), e1 = __expf(s1 - m), e2 = __expf(s2 - m);
  float ss = e0 + e1 + e2;
  long idx = (long)b * NB + c;
  float v = (e0 * a_ts[idx] + e1 * a_di[idx] + e2 * a_de[idx]) / ss;
  adjF[idx] = v; adjF16[idx] = (_Float16)v;
}

// ---------------- MPGRU prep buffers ----------------
__global__ void k_pack_delta(const float* delta, _Float16* dt){
  long i = (long)blockIdx.x * blockDim.x + threadIdx.x; // over [b][t][c]
  if (i >= (long)NB * NT * DDM) return;
  int c = (int)(i % DDM);
  long bt = i / DDM;
  int t = (int)(bt % NT);
  int b = (int)(bt / NT);
  dt[((long)t * DDM + c) * NB + b] = (_Float16)delta[i];
}
__global__ void k_build_inpt(const float* gru_out, const float* x,
                             const float* diag, const float* demo,
                             _Float16* inp16, unsigned char* maskb){
  long i = (long)blockIdx.x * blockDim.x + threadIdx.x; // over [t][c][b]
  if (i >= (long)NT * DDM * NB) return;
  int b = (int)(i % NB);
  long tc = i / NB;
  int c = (int)(tc % DDM);
  int t = (int)(tc / DDM);
  float v, mref;
  if (c < XDM){
    v = gru_out[((long)b * NT + t) * XDM + c];
    mref = x[((long)b * NT + t) * XDM + c];
  } else if (c < XDM + DGD){
    v = diag[(long)b * DGD + (c - XDM)]; mref = v;
  } else {
    v = demo[(long)b * DMD + (c - XDM - DGD)]; mref = v;
  }
  inp16[i] = (_Float16)v;
  maskb[i] = (mref != -1.0f) ? 1 : 0;
}
// Ghat[2048,704]: rows k*512+i = [Wr;Wu] row i, cols k*704..k*704+703 of GIN
__global__ void k_build_ghat(const float* Wr, const float* Wu, _Float16* ghat){
  long i = (long)blockIdx.x * blockDim.x + threadIdx.x;
  if (i >= (long)2048 * C0) return;
  int j = (int)(i % C0);
  int row = (int)(i / C0);
  int k = row >> 9, ii = row & 511;
  const float* W = (ii < 256) ? Wr : Wu;
  int rr = ii & 255;
  ghat[i] = (_Float16)W[(long)rr * GIN + k * C0 + j];
}
__global__ void k_build_wchat(const float* Wc, _Float16* wch){
  long i = (long)blockIdx.x * blockDim.x + threadIdx.x;
  if (i >= (long)1024 * C0) return;
  int j = (int)(i % C0);
  int row = (int)(i / C0);
  int k = row >> 8, rr = row & 255;
  wch[i] = (_Float16)Wc[(long)rr * GIN + k * C0 + j];
}

// ---------------- MPGRU per-step kernels ----------------
// K1: x_hat = predW@h + b (WMMA, 32x32/wave) -> out, Z0 rows [x_s; mask; g*h], H0 = g*h
__global__ void k_step_pre(const _Float16* __restrict__ predW16, const float* __restrict__ predb,
                           const _Float16* __restrict__ h16, const float* __restrict__ hf32,
                           const _Float16* __restrict__ inp16, const unsigned char* __restrict__ maskb,
                           const _Float16* __restrict__ gamma16,
                           float* __restrict__ out, _Float16* __restrict__ Z0,
                           float* __restrict__ H0, int t){
  int lane = threadIdx.x & 31;
  int wave = threadIdx.x >> 5;
  int tile = blockIdx.x * (blockDim.x >> 5) + wave;
  const int gemmTiles = 7 * 8;       // M=224, N=256, 32x32 macro-tiles
  if (tile < gemmTiles){
    int m0 = (tile >> 3) << 5;
    int n0 = (tile & 7) << 5;
    const _Float16* Ap0 = predW16 + (long)m0 * HID;
    const _Float16* Ap1 = Ap0 + (long)16 * HID;
    v8f cc[2][2] = {{{}, {}}, {{}, {}}};
    for (int k0 = 0; k0 < HID; k0 += 32){
      v16h a0 = load_a16(Ap0 + k0, HID, lane);
      v16h a1 = load_a16(Ap1 + k0, HID, lane);
      const _Float16* bp = h16 + (long)k0 * NB + n0;
      v16h b0 = load_b16(bp,      NB, lane);
      v16h b1 = load_b16(bp + 16, NB, lane);
      cc[0][0] = wmma16(a0, b0, cc[0][0]);
      cc[0][1] = wmma16(a0, b1, cc[0][1]);
      cc[1][0] = wmma16(a1, b0, cc[1][0]);
      cc[1][1] = wmma16(a1, b1, cc[1][1]);
    }
    long tb = (long)t * DDM * NB;
#pragma unroll
    for (int ti = 0; ti < 2; ti++)
#pragma unroll
      for (int tj = 0; tj < 2; tj++){
        int mrow = m0 + ti*16 + 8*(lane >> 4);
        int n    = n0 + tj*16 + (lane & 15);
#pragma unroll
        for (int r = 0; r < 8; r++){
          int c = mrow + r;
          float xh = cc[ti][tj][r] + predb[c];
          out[((long)n * NT + t) * DDM + c] = xh;
          int mi = maskb[tb + (long)c * NB + n];
          float xs = mi ? (float)inp16[tb + (long)c * NB + n] : xh;
          Z0[(long)c * NB + n] = (_Float16)xs;
          Z0[(long)(DDM + c) * NB + n] = (_Float16)(mi ? 1.0f : 0.0f);
        }
      }
  } else {
    int tt = tile - gemmTiles;       // 256 decay tiles, 256 elems each
    if (tt >= 256) return;
    int base = tt * 256 + lane * 8;
    long gb = (long)t * HID * NB;
#pragma unroll
    for (int e = 0; e < 8; e++){
      int idx = base + e;            // over HID*NB
      float h0v = (float)gamma16[gb + idx] * hf32[idx];
      H0[idx] = h0v;
      Z0[(long)(2 * DDM) * NB + idx] = (_Float16)h0v;
    }
  }
}

// K2b: pre = Y0 + sum_k Yk@AkT + b -> r (writes RH into Z0 h-rows) and u
__global__ void k_gates_ru(const _Float16* __restrict__ Y,
                           const _Float16* __restrict__ A1T, const _Float16* __restrict__ A2T,
                           const _Float16* __restrict__ A3T,
                           const float* __restrict__ br, const float* __restrict__ bu,
                           const float* __restrict__ H0, float* __restrict__ U,
                           _Float16* __restrict__ Z0){
  int lane = threadIdx.x & 31;
  int wave = threadIdx.x >> 5;
  int tile = blockIdx.x * (blockDim.x >> 5) + wave;
  if (tile >= 16 * 8) return;        // M=512, N=256, 32x32 macro-tiles
  int m0 = (tile >> 3) << 5;
  int n0 = (tile & 7) << 5;
  const _Float16* mats[3] = {A1T, A2T, A3T};
  v8f cc[2][2] = {{{}, {}}, {{}, {}}};
#pragma unroll
  for (int k = 1; k <= 3; k++){
    const _Float16* Ak = mats[k - 1];
    const _Float16* Yk0 = Y + (long)(512 * k + m0) * NB;
    const _Float16* Yk1 = Yk0 + (long)16 * NB;
    for (int k0 = 0; k0 < NB; k0 += 32){
      v16h a0 = load_a16(Yk0 + k0, NB, lane);
      v16h a1 = load_a16(Yk1 + k0, NB, lane);
      const _Float16* bp = Ak + (long)k0 * NB + n0;
      v16h b0 = load_b16(bp,      NB, lane);
      v16h b1 = load_b16(bp + 16, NB, lane);
      cc[0][0] = wmma16(a0, b0, cc[0][0]);
      cc[0][1] = wmma16(a0, b1, cc[0][1]);
      cc[1][0] = wmma16(a1, b0, cc[1][0]);
      cc[1][1] = wmma16(a1, b1, cc[1][1]);
    }
  }
#pragma unroll
  for (int ti = 0; ti < 2; ti++)
#pragma unroll
    for (int tj = 0; tj < 2; tj++){
      int mrow = m0 + ti*16 + 8*(lane >> 4);
      int n    = n0 + tj*16 + (lane & 15);
#pragma unroll
      for (int r = 0; r < 8; r++){
        int i = mrow + r;
        float v = cc[ti][tj][r] + (float)Y[(long)i * NB + n];
        if (i < 256){
          float rv = sigf(v + br[i]);
          Z0[(long)(2 * DDM + i) * NB + n] = (_Float16)(rv * H0[(long)i * NB + n]);
        } else {
          int iu = i - 256;
          U[(long)iu * NB + n] = sigf(v + bu[iu]);
        }
      }
    }
}

// K4: c = tanh(Yc0 + sum Yck@AkT + bc); h = u*H0 + (1-u)*c
__global__ void k_gates_c(const _Float16* __restrict__ Yc,
                          const _Float16* __restrict__ A1T, const _Float16* __restrict__ A2T,
                          const _Float16* __restrict__ A3T,
                          const float* __restrict__ bc, const float* __restrict__ U,
                          const float* __restrict__ H0,
                          float* __restrict__ hf32, _Float16* __restrict__ h16){
  int lane = threadIdx.x & 31;
  int wave = threadIdx.x >> 5;
  int tile = blockIdx.x * (blockDim.x >> 5) + wave;
  if (tile >= 8 * 8) return;         // M=256, N=256, 32x32 macro-tiles
  int m0 = (tile >> 3) << 5;
  int n0 = (tile & 7) << 5;
  const _Float16* mats[3] = {A1T, A2T, A3T};
  v8f cc[2][2] = {{{}, {}}, {{}, {}}};
#pragma unroll
  for (int k = 1; k <= 3; k++){
    const _Float16* Ak = mats[k - 1];
    const _Float16* Yk0 = Yc + (long)(256 * k + m0) * NB;
    const _Float16* Yk1 = Yk0 + (long)16 * NB;
    for (int k0 = 0; k0 < NB; k0 += 32){
      v16h a0 = load_a16(Yk0 + k0, NB, lane);
      v16h a1 = load_a16(Yk1 + k0, NB, lane);
      const _Float16* bp = Ak + (long)k0 * NB + n0;
      v16h b0 = load_b16(bp,      NB, lane);
      v16h b1 = load_b16(bp + 16, NB, lane);
      cc[0][0] = wmma16(a0, b0, cc[0][0]);
      cc[0][1] = wmma16(a0, b1, cc[0][1]);
      cc[1][0] = wmma16(a1, b0, cc[1][0]);
      cc[1][1] = wmma16(a1, b1, cc[1][1]);
    }
  }
#pragma unroll
  for (int ti = 0; ti < 2; ti++)
#pragma unroll
    for (int tj = 0; tj < 2; tj++){
      int mrow = m0 + ti*16 + 8*(lane >> 4);
      int n    = n0 + tj*16 + (lane & 15);
#pragma unroll
      for (int r = 0; r < 8; r++){
        int i = mrow + r;
        float v = cc[ti][tj][r] + (float)Yc[(long)i * NB + n] + bc[i];
        float cv = tanhf(v);
        float u = U[(long)i * NB + n];
        float hn = u * H0[(long)i * NB + n] + (1.0f - u) * cv;
        hf32[(long)i * NB + n] = hn;
        h16 [(long)i * NB + n] = (_Float16)hn;
      }
    }
}

} // namespace

extern "C" void kernel_launch(void* const* d_in, const int* in_sizes, int n_in,
                              void* d_out, int out_size, void* d_ws, size_t ws_size,
                              hipStream_t stream) {
  (void)in_sizes; (void)n_in; (void)out_size; (void)ws_size;
  // -------- inputs --------
  const float* x     = (const float*)d_in[0];
  const float* diag  = (const float*)d_in[1];
  const float* demo  = (const float*)d_in[2];
  const float* delta = (const float*)d_in[3];
  const int*   lens  = (const int*)d_in[4];
  const float* h0g   = (const float*)d_in[5];
  const float* gWih  = (const float*)d_in[6];
  const float* gbih  = (const float*)d_in[7];
  const float* gWhh  = (const float*)d_in[8];
  const float* gbhh  = (const float*)d_in[9];
  const float* tsWq  = (const float*)d_in[10]; const float* tsbq = (const float*)d_in[11];
  const float* tsWk  = (const float*)d_in[12]; const float* tsbk = (const float*)d_in[13];
  const float* tsWo  = (const float*)d_in[14]; const float* tsbo = (const float*)d_in[15];
  const float* diWq  = (const float*)d_in[16]; const float* dibq = (const float*)d_in[17];
  const float* diWk  = (const float*)d_in[18]; const float* dibk = (const float*)d_in[19];
  const float* diWo  = (const float*)d_in[20]; const float* dibo = (const float*)d_in[21];
  const float* deWq  = (const float*)d_in[22]; const float* debq = (const float*)d_in[23];
  const float* deWk  = (const float*)d_in[24]; const float* debk = (const float*)d_in[25];
  const float* deWo  = (const float*)d_in[26]; const float* debo = (const float*)d_in[27];
  const float* fWts  = (const float*)d_in[28]; const float* fbts = (const float*)d_in[29];
  const float* fWdi  = (const float*)d_in[30]; const float* fbdi = (const float*)d_in[31];
  const float* fWde  = (const float*)d_in[32]; const float* fbde = (const float*)d_in[33];
  const float* fWall = (const float*)d_in[34]; const float* fball= (const float*)d_in[35];
  const float* cWr   = (const float*)d_in[36]; const float* cbr  = (const float*)d_in[37];
  const float* cWu   = (const float*)d_in[38]; const float* cbu  = (const float*)d_in[39];
  const float* cWc   = (const float*)d_in[40]; const float* cbc  = (const float*)d_in[41];
  const float* predW = (const float*)d_in[42]; const float* predb= (const float*)d_in[43];
  const float* wdgW  = (const float*)d_in[44]; const float* wdgb = (const float*)d_in[45];
  float* out = (float*)d_out;

  // -------- workspace carve-up --------
  char* ws = (char*)d_ws;
  size_t cur = 0;
  auto alloc = [&](size_t bytes)->char*{
    cur = (cur + 255) & ~(size_t)255;
    char* p = ws + cur; cur += bytes; return p;
  };
  _Float16* xf16   = (_Float16*)alloc((size_t)NB*NT*XDM*2);
  float*    t_mean = (float*)   alloc((size_t)NB*XDM*4);
  float*    s_acc  = (float*)   alloc((size_t)2*XDM*4);
  _Float16* WihT   = (_Float16*)alloc((size_t)XDM*384*2);
  _Float16* WhhT   = (_Float16*)alloc((size_t)XDM*384*2);
  _Float16* tsWqT  = (_Float16*)alloc((size_t)128*128*2);
  _Float16* tsWkT  = (_Float16*)alloc((size_t)128*128*2);
  _Float16* diWqT  = (_Float16*)alloc((size_t)64*64*2);
  _Float16* diWkT  = (_Float16*)alloc((size_t)64*64*2);
  _Float16* deWqT  = (_Float16*)alloc((size_t)32*32*2);
  _Float16* deWkT  = (_Float16*)alloc((size_t)32*32*2);
  _Float16* fWtsT  = (_Float16*)alloc((size_t)256*128*2);
  _Float16* fWdiT  = (_Float16*)alloc((size_t)256*128*2);
  _Float16* fWdeT  = (_Float16*)alloc((size_t)256*128*2);
  _Float16* fWallT = (_Float16*)alloc((size_t)768*128*2);
  _Float16* predW16= (_Float16*)alloc((size_t)DDM*HID*2);
  _Float16* wdgW16 = (_Float16*)alloc((size_t)HID*DDM*2);
  _Float16* ghat   = (_Float16*)alloc((size_t)2048*C0*2);
  _Float16* wchat  = (_Float16*)alloc((size_t)1024*C0*2);
  _Float16* diag16 = (_Float16*)alloc((size_t)NB*DGD*2);
  _Float16* demo16 = (_Float16*)alloc((size_t)NB*DMD*2);
  float*    gruOut = (float*)   alloc((size_t)NB*NT*XDM*4);
  _Float16* last16 = (_Float16*)alloc((size_t)NB*XDM*2);
  float*    qbuf   = (float*)   alloc((size_t)NB*128*4);
  float*    kbuf   = (float*)   alloc((size_t)NB*128*4);
  float*    a_ts   = (float*)   alloc((size_t)NB*NB*4);
  float*    a_di   = (float*)   alloc((size_t)NB*NB*4);
  float*    a_de   = (float*)   alloc((size_t)NB*NB*4);
  _Float16* adjall = (_Float16*)alloc((size_t)NB*768*2);
  float*    q_all  = (float*)   alloc((size_t)NB*128*4);
  float*    q_tsb  = (float*)   alloc((size_t)NB*128*4);
  float*    q_dib  = (float*)   alloc((size_t)NB*128*4);
  float*    q_deb  = (float*)   alloc((size_t)NB*128*4);
  float*    adjF   = (float*)   alloc((size_t)NB*NB*4);
  _Float16* adjF16 = (_Float16*)alloc((size_t)NB*NB*2);
  float*    A2f    = (float*)   alloc((size_t)NB*NB*4);
  _Float16* A2h    = (_Float16*)alloc((size_t)NB*NB*2);
  float*    A3f    = (float*)   alloc((size_t)NB*NB*4);
  _Float16* A1T    = (_Float16*)alloc((size_t)NB*NB*2);
  _Float16* A2T    = (_Float16*)alloc((size_t)NB*NB*2);
  _Float16* A3T    = (_Float16*)alloc((size_t)NB*NB*2);
  _Float16* deltaT = (_Float16*)alloc((size_t)NT*DDM*NB*2);
  _Float16* gamma16= (_Float16*)alloc((size_t)NT*HID*NB*2);
  _Float16* inp16  = (_Float16*)alloc((size_t)NT*DDM*NB*2);
  unsigned char* maskb = (unsigned char*)alloc((size_t)NT*DDM*NB);
  float*    hf32   = (float*)   alloc((size_t)HID*NB*4);
  _Float16* h16    = (_Float16*)alloc((size_t)HID*NB*2);
  _Float16* Z0     = (_Float16*)alloc((size_t)C0*NB*2);
  float*    H0     = (float*)   alloc((size_t)HID*NB*4);
  _Float16* Ybuf   = (_Float16*)alloc((size_t)2048*NB*2);
  float*    Ubuf   = (float*)   alloc((size_t)HID*NB*4);
  _Float16* Ycbuf  = (_Float16*)alloc((size_t)1024*NB*2);

  auto gemm = [&](const _Float16* A, int lda, const _Float16* Bm, int ldb,
                  float* Cf, _Float16* Ch, int ldc, int M, int N, int K,
                  const float* bias, int biasMode, int act,
                  int batch, long sB, long sC){
    int tiles = (M / 32) * (N / 32);
    dim3 grid((tiles + 7) / 8, batch);
    k_gemm<<<grid, 256, 0, stream>>>(A, lda, Bm, ldb, Cf, Ch, ldc, M, N, K,
                                     bias, biasMode, act, sB, sC);
  };
  auto g1d = [&](long n){ return dim3((unsigned)((n + 255) / 256)); };

  // -------- phase A: imputation + weight prep --------
  k_zero<<<1, 256, 0, stream>>>((unsigned int*)s_acc, 256);
  k_imp_stats<<<NB, XDM, 0, stream>>>(x, t_mean, s_acc);
  { long n = (long)NB*NT*XDM; k_imp_fill<<<g1d(n), 256, 0, stream>>>(x, t_mean, s_acc, xf16); }

  { long n=(long)384*XDM; k_t16<<<g1d(n),256,0,stream>>>(gWih, WihT, 384, XDM);
                          k_t16<<<g1d(n),256,0,stream>>>(gWhh, WhhT, 384, XDM); }
  { long n=128L*128; k_t16<<<g1d(n),256,0,stream>>>(tsWq, tsWqT, 128, 128);
                     k_t16<<<g1d(n),256,0,stream>>>(tsWk, tsWkT, 128, 128); }
  { long n=64L*64;   k_t16<<<g1d(n),256,0,stream>>>(diWq, diWqT, 64, 64);
                     k_t16<<<g1d(n),256,0,stream>>>(diWk, diWkT, 64, 64); }
  { long n=32L*32;   k_t16<<<g1d(n),256,0,stream>>>(deWq, deWqT, 32, 32);
                     k_t16<<<g1d(n),256,0,stream>>>(deWk, deWkT, 32, 32); }
  { long n=128L*256; k_t16<<<g1d(n),256,0,stream>>>(fWts, fWtsT, 128, 256);
                     k_t16<<<g1d(n),256,0,stream>>>(fWdi, fWdiT, 128, 256);
                     k_t16<<<g1d(n),256,0,stream>>>(fWde, fWdeT, 128, 256); }
  { long n=128L*768; k_t16<<<g1d(n),256,0,stream>>>(fWall, fWallT, 128, 768); }
  { long n=(long)DDM*HID; k_c16<<<g1d(n),256,0,stream>>>(predW, predW16, n); }
  { long n=(long)HID*DDM; k_c16<<<g1d(n),256,0,stream>>>(wdgW,  wdgW16,  n); }
  { long n=(long)NB*DGD;  k_c16<<<g1d(n),256,0,stream>>>(diag,  diag16,  n); }
  { long n=(long)NB*DMD;  k_c16<<<g1d(n),256,0,stream>>>(demo,  demo16,  n); }
  { long n=2048L*C0; k_build_ghat<<<g1d(n),256,0,stream>>>(cWr, cWu, ghat); }
  { long n=1024L*C0; k_build_wchat<<<g1d(n),256,0,stream>>>(cWc, wchat); }

  // -------- phase B: GRU + adjacency construction --------
  k_gru<<<NB/16, 256, 0, stream>>>(xf16, WihT, WhhT, gbih, gbhh, h0g, gruOut);
  k_last<<<NB, XDM, 0, stream>>>(gruOut, lens, last16);

  gemm(last16, XDM, tsWqT, 128, qbuf, nullptr, 128, NB, 128, 128, tsbq, 2, 0, 1, 0, 0);
  gemm(last16, XDM, tsWkT, 128, kbuf, nullptr, 128, NB, 128, 128, tsbk, 2, 0, 1, 0, 0);
  k_attn<<<NB, 256, 0, stream>>>(qbuf, kbuf, tsWo, tsbo, 128, 8, a_ts, adjall, 0);
  gemm(diag16, DGD, diWqT, 64, qbuf, nullptr, 64, NB, 64, 64, dibq, 2, 0, 1, 0, 0);
  gemm(diag16, DGD, diWkT, 64, kbuf, nullptr, 64, NB, 64, 64, dibk, 2, 0, 1, 0, 0);
  k_attn<<<NB, 256, 0, stream>>>(qbuf, kbuf, diWo, dibo, 64, 8, a_di, adjall, 256);
  gemm(demo16, DMD, deWqT, 32, qbuf, nullptr, 32, NB, 32, 32, debq, 2, 0, 1, 0, 0);
  gemm(demo16, DMD, deWkT, 32, kbuf, nullptr, 32, NB, 32, 32, debk, 2, 0, 1, 0, 0);
  k_attn<<<NB, 256, 0, stream>>>(qbuf, kbuf, deWo, debo, 32, 4, a_de, adjall, 512);

  gemm(adjall,       768, fWallT, 128, q_all, nullptr, 128, NB, 128, 768, fball, 2, 0, 1, 0, 0);
  gemm(adjall + 0,   768, fWtsT,  128, q_tsb, nullptr, 128, NB, 128, 256, fbts,  2, 0, 1, 0, 0);
  gemm(adjall + 256, 768, fWdiT,  128, q_dib, nullptr, 128, NB, 128, 256, fbdi,  2, 0, 1, 0, 0);
  gemm(adjall + 512, 768, fWdeT,  128, q_deb, nullptr, 128, NB, 128, 256, fbde,  2, 0, 1, 0, 0);
  k_fuse<<<NB, 256, 0, stream>>>(q_all, q_tsb, q_dib, q_deb, a_ts, a_di, a_de, adjF, adjF16);

  gemm(adjF16, NB, adjF16, NB, A2f, nullptr, NB, NB, NB, NB, nullptr, 0, 0, 1, 0, 0);
  { long n=(long)NB*NB; k_c16<<<g1d(n),256,0,stream>>>(A2f, A2h, n); }
  gemm(A2h, NB, adjF16, NB, A3f, nullptr, NB, NB, NB, NB, nullptr, 0, 0, 1, 0, 0);
  { long n=(long)NB*NB; k_t16<<<g1d(n),256,0,stream>>>(adjF, A1T, NB, NB);
                        k_t16<<<g1d(n),256,0,stream>>>(A2f,  A2T, NB, NB);
                        k_t16<<<g1d(n),256,0,stream>>>(A3f,  A3T, NB, NB); }

  // -------- phase C: MPGRU prep --------
  { long n=(long)NB*NT*DDM; k_pack_delta<<<g1d(n),256,0,stream>>>(delta, deltaT); }
  gemm(wdgW16, DDM, deltaT, NB, nullptr, gamma16, NB, HID, NB, DDM,
       wdgb, 1, 3, NT, (long)DDM*NB, (long)HID*NB);
  { long n=(long)NT*DDM*NB; k_build_inpt<<<g1d(n),256,0,stream>>>(gruOut, x, diag, demo, inp16, maskb); }
  k_zero<<<g1d((long)HID*NB), 256, 0, stream>>>((unsigned int*)hf32, (long)HID*NB);
  k_zero<<<g1d((long)HID*NB/2), 256, 0, stream>>>((unsigned int*)h16, (long)HID*NB/2);

  // -------- phase D: MPGRU time loop --------
  for (int t = 0; t < NT; t++){
    k_step_pre<<<39, 256, 0, stream>>>(predW16, predb, h16, hf32, inp16, maskb,
                                       gamma16, out, Z0, H0, t);
    k_gemm_big<<<dim3(8, 8), 256, 0, stream>>>(ghat,  Z0, Ybuf);   // [2048,704]x[704,256]
    k_gates_ru<<<16, 256, 0, stream>>>(Ybuf, A1T, A2T, A3T, cbr, cbu, H0, Ubuf, Z0);
    k_gemm_big<<<dim3(4, 8), 256, 0, stream>>>(wchat, Z0, Ycbuf);  // [1024,704]x[704,256]
    k_gates_c<<<8, 256, 0, stream>>>(Ycbuf, A1T, A2T, A3T, cbc, Ubuf, H0, hf32, h16);
  }
}
